// TransformerBlock_84404697301528
// MI455X (gfx1250) — matrix operand order repaired and checked by reading, not compiled, not run on hardware
//
#include <hip/hip_runtime.h>
#include <hip/hip_bf16.h>

// Problem constants (match reference)
#define DIMC 1024
#define NHEAD 16
#define HDIM 64
#define TSEQ 2048
#define BSZ 4
#define MROWS (BSZ * TSEQ)   // 8192
#define FFDIM 4096

typedef __bf16 bf16_t;
typedef __attribute__((ext_vector_type(16))) __bf16 v16bf;
typedef __attribute__((ext_vector_type(8)))  __bf16 v8bf;
typedef __attribute__((ext_vector_type(8)))  float  v8f;

// ---------------------------------------------------------------------------
// WMMA helper: D = A(16x32 bf16) * B(32x16 bf16) + C(16x16 f32)
// ---------------------------------------------------------------------------
static __device__ __forceinline__ v8f wmma_bf16(v16bf a, v16bf b, v8f c) {
  return __builtin_amdgcn_wmma_f32_16x16x32_bf16(
      /*neg_a=*/false, a, /*neg_b=*/false, b,
      /*c_mod=*/(short)0, c, /*reuse_a=*/false, /*reuse_b=*/false);
}

static __device__ __forceinline__ v16bf cat16(v8bf lo, v8bf hi) {
  return __builtin_shufflevector(lo, hi, 0, 1, 2, 3, 4, 5, 6, 7,
                                 8, 9, 10, 11, 12, 13, 14, 15);
}

// ---------------------------------------------------------------------------
// CDNA5 async global->LDS copy (16B granule) + asynccnt wait, with fallback
// Builtin signature (from compiler diagnostic): param0 = v4i addrspace(1)*
// ---------------------------------------------------------------------------
typedef int v4i_g __attribute__((vector_size(16)));
typedef __attribute__((address_space(1))) v4i_g* as1_v4i_ptr;
typedef __attribute__((address_space(3))) v4i_g* as3_v4i_ptr;

static __device__ __forceinline__ void async_copy16(const bf16_t* g, bf16_t* l) {
#if __has_builtin(__builtin_amdgcn_global_load_async_to_lds_b128)
  __builtin_amdgcn_global_load_async_to_lds_b128(
      (as1_v4i_ptr)g, (as3_v4i_ptr)l, 0, 0);
#else
  *(v8bf*)l = *(const v8bf*)g;
#endif
}

static __device__ __forceinline__ void wait_async0() {
#if __has_builtin(__builtin_amdgcn_global_load_async_to_lds_b128)
#if __has_builtin(__builtin_amdgcn_s_wait_asynccnt)
  __builtin_amdgcn_s_wait_asynccnt(0);
#else
  asm volatile("s_wait_asynccnt 0x0" ::: "memory");
#endif
#endif
}

// A-fragment (16x32 bf16): per lane, one contiguous 32B run of
// row `row`: k = h*16 + [0,16)  ->  2x b128 loads.
static __device__ __forceinline__ v16bf load_a_frag(const bf16_t* __restrict__ A,
                                                    int lda, int row, int k0, int lane) {
  const bf16_t* p = A + (size_t)row * lda + k0 + ((lane >> 4) << 4);
  v8bf lo = *(const v8bf*)(p);
  v8bf hi = *(const v8bf*)(p + 8);
  return cat16(lo, hi);
}

// B-fragment (32x16 bf16) from K-major storage BT[n][k] (contiguous along K):
// per lane one contiguous 32B run: k = h*16 + [0,16)  ->  2x b128 loads.
static __device__ __forceinline__ v16bf load_bk_frag(const bf16_t* __restrict__ BT,
                                                     int ldn, int k0, int n0, int lane) {
  const bf16_t* p = BT + (size_t)(n0 + (lane & 15)) * ldn + k0 + ((lane >> 4) << 4);
  v8bf lo = *(const v8bf*)(p);
  v8bf hi = *(const v8bf*)(p + 8);
  return cat16(lo, hi);
}

// B-fragment from an LDS tile stored [16+ cols][40 bf16] (80B row stride).
static __device__ __forceinline__ v16bf load_bk_lds(const bf16_t* Bt, int lane) {
  const bf16_t* p = Bt + (size_t)(lane & 15) * 40 + ((lane >> 4) << 4);
  v8bf lo = *(const v8bf*)(p);
  v8bf hi = *(const v8bf*)(p + 8);
  return cat16(lo, hi);
}

// ---------------------------------------------------------------------------
// Weight prep: fp32 W[K,N] -> bf16 W^T[N,K] (K-major for contiguous B-frags)
// ---------------------------------------------------------------------------
__global__ void k_conv_transpose(const float* __restrict__ W, bf16_t* __restrict__ WT,
                                 int K, int N) {
  int i = blockIdx.x * blockDim.x + threadIdx.x;
  if (i >= K * N) return;
  int k = i / N, n = i - k * N;
  WT[(size_t)n * K + k] = (bf16_t)W[i];
}

// ---------------------------------------------------------------------------
// RMSNorm: out_bf16[row] = x[row] / sqrt(mean(x^2)+eps) * g
// ---------------------------------------------------------------------------
__global__ void k_rmsnorm(const float* __restrict__ X, const float* __restrict__ g,
                          bf16_t* __restrict__ out) {
  const int C = DIMC;
  const int row = blockIdx.x;
  const float* x = X + (size_t)row * C;
  float ss = 0.f;
  for (int c = threadIdx.x; c < C; c += 256) { float v = x[c]; ss += v * v; }
#pragma unroll
  for (int off = 16; off > 0; off >>= 1) ss += __shfl_xor(ss, off, 32);
  __shared__ float red[8];
  const int lane = threadIdx.x & 31, wv = threadIdx.x >> 5;
  if (lane == 0) red[wv] = ss;
  __syncthreads();
  if (threadIdx.x == 0) {
    float t = 0.f;
#pragma unroll
    for (int i = 0; i < 8; ++i) t += red[i];
    red[0] = rsqrtf(t / (float)C + 1e-6f);
  }
  __syncthreads();
  const float inv = red[0];
  bf16_t* o = out + (size_t)row * C;
  for (int c = threadIdx.x; c < C; c += 256) o[c] = (bf16_t)(x[c] * inv * g[c]);
}

// ---------------------------------------------------------------------------
// bf16 WMMA GEMM: acc = A[M,K] @ WT[N,K]^T
// 256 threads = 8 waves, block tile 128(M) x 64(N), K-step 32.
// B tile (64x32) is double-buffered through LDS via async global->LDS loads
// (shared by all 8 waves); A fragments are direct contiguous b128 loads.
// MODE 0: OUT bf16 [M,N]
// MODE 1: OUT f32  [M,N] = RES + acc
// MODE 2: OUT bf16 scattered into [B,NH,HD,T] (transposed V for attention)
// ---------------------------------------------------------------------------
template <int MODE>
__global__ void __launch_bounds__(256) k_gemm(const bf16_t* __restrict__ A,
                                              const bf16_t* __restrict__ WT,
                                              const float* __restrict__ RES,
                                              void* __restrict__ OUT,
                                              int M, int N, int K) {
  const int lane = threadIdx.x & 31;
  const int wave = threadIdx.x >> 5;
  const int m0 = blockIdx.x * 128 + wave * 16;
  const int n0 = blockIdx.y * 64;
  const int row = m0 + (lane & 15);

  // double-buffered B tile: [buf][col 0..63][40 bf16] (80B stride, 16B aligned)
  __shared__ __align__(16) bf16_t Bt[2][64][40];

  // stage one 64x32 B tile: 256 threads x one 16B chunk each
  const int scol = threadIdx.x >> 2;
  const int skp = (threadIdx.x & 3) << 3;
  const bf16_t* sp = WT + (size_t)(n0 + scol) * K + skp;

  v8f acc[4] = {};
  async_copy16(sp, &Bt[0][scol][skp]);
  int cur = 0;
  for (int k0 = 0; k0 < K; k0 += 32) {
    wait_async0();
    __syncthreads();  // staged tile visible to all waves
    if (k0 + 32 < K) async_copy16(sp + k0 + 32, &Bt[cur ^ 1][scol][skp]);
    v16bf a = load_a_frag(A, K, row, k0, lane);
#pragma unroll
    for (int c = 0; c < 4; ++c) {
      v16bf b = load_bk_lds(&Bt[cur][c * 16][0], lane);
      acc[c] = wmma_bf16(a, b, acc[c]);
    }
    __syncthreads();  // all reads of `cur` done before it is refilled
    cur ^= 1;
  }

  const int half = lane >> 4, col = lane & 15;
#pragma unroll
  for (int c = 0; c < 4; ++c)
#pragma unroll
    for (int r = 0; r < 8; ++r) {
      const size_t m = m0 + r + 8 * half;
      const size_t n = n0 + 16 * c + col;
      if (MODE == 0) {
        ((bf16_t*)OUT)[m * N + n] = (bf16_t)acc[c][r];
      } else if (MODE == 1) {
        ((float*)OUT)[m * N + n] = acc[c][r] + RES[m * N + n];
      } else {
        const size_t bb = m / TSEQ, tt = m - bb * TSEQ;
        const size_t hh = n >> 6, dd = n & 63;
        ((bf16_t*)OUT)[(((bb * NHEAD) + hh) * HDIM + dd) * TSEQ + tt] =
            (bf16_t)acc[c][r];
      }
    }
}

// ---------------------------------------------------------------------------
// Fused SwiGLU GEMM: OUT_bf16 = silu(A@W1) * (A@W3), shared A-fragments,
// direct contiguous global fragment loads.
// ---------------------------------------------------------------------------
__global__ void k_gemm_swiglu(const bf16_t* __restrict__ A, const bf16_t* __restrict__ W1T,
                              const bf16_t* __restrict__ W3T, bf16_t* __restrict__ OUT,
                              int M, int N, int K) {
  const int lane = threadIdx.x & 31;
  const int wave = threadIdx.x >> 5;
  const int m0 = blockIdx.x * 64 + wave * 16;
  const int n0 = blockIdx.y * 64;
  const int row = m0 + (lane & 15);
  v8f acc1[4] = {}, acc3[4] = {};
  for (int k0 = 0; k0 < K; k0 += 32) {
    if (k0 + 32 < K) __builtin_prefetch(A + (size_t)row * K + k0 + 32, 0, 1);
    v16bf a = load_a_frag(A, K, row, k0, lane);
#pragma unroll
    for (int c = 0; c < 4; ++c) {
      v16bf b1 = load_bk_frag(W1T, K, k0, n0 + 16 * c, lane);
      acc1[c] = wmma_bf16(a, b1, acc1[c]);
      v16bf b3 = load_bk_frag(W3T, K, k0, n0 + 16 * c, lane);
      acc3[c] = wmma_bf16(a, b3, acc3[c]);
    }
  }
  const int half = lane >> 4, col = lane & 15;
#pragma unroll
  for (int c = 0; c < 4; ++c)
#pragma unroll
    for (int r = 0; r < 8; ++r) {
      float u = acc1[c][r];
      float t = acc3[c][r];
      float gsi = u / (1.f + __expf(-u));  // silu
      OUT[(size_t)(m0 + r + 8 * half) * N + n0 + 16 * c + col] = (bf16_t)(gsi * t);
    }
}

// ---------------------------------------------------------------------------
// RoPE in-place on Q and K (layout [B,T,NH,HD] flattened = [B,T,C]).
// One thread per rotation pair (d, d+32); cos/sin equal for both halves.
// ---------------------------------------------------------------------------
__global__ void k_rope(bf16_t* __restrict__ Q, bf16_t* __restrict__ K) {
  const int idx = blockIdx.x * blockDim.x + threadIdx.x;  // over B*T*NH*32
  const int j = idx & 31;
  const int h = (idx >> 5) & (NHEAD - 1);
  const int t = (idx >> 9) & (TSEQ - 1);
  const int b = idx >> 20;
  if (b >= BSZ) return;
  // inv_freq = 10000^(-2j/HD); ln(10000) = 9.210340371976184
  const float ang = (float)t * __expf(-(float)(2 * j) / (float)HDIM * 9.210340371976184f);
  const float c = __cosf(ang), s = __sinf(ang);
  const size_t base = ((size_t)(b * TSEQ + t) * DIMC) + h * HDIM;
  {
    float lo = (float)Q[base + j], hi = (float)Q[base + j + 32];
    Q[base + j]      = (bf16_t)(lo * c - hi * s);
    Q[base + j + 32] = (bf16_t)(hi * c + lo * s);
  }
  {
    float lo = (float)K[base + j], hi = (float)K[base + j + 32];
    K[base + j]      = (bf16_t)(lo * c - hi * s);
    K[base + j + 32] = (bf16_t)(hi * c + lo * s);
  }
}

// ---------------------------------------------------------------------------
// Flash attention: one wave per (b, h, 16-row Q tile). Online softmax.
// Q,K fragments contiguous from [B,T,C]; V fragments contiguous from the
// transposed [B,NH,HD,T] layout; P goes via an LDS tile (padded rows: 40
// bf16 = 80B stride -> conflict-free, 16B-aligned ds_load_b128).
// ---------------------------------------------------------------------------
__global__ void __launch_bounds__(32) k_attn(const bf16_t* __restrict__ Q,
                                             const bf16_t* __restrict__ K,
                                             const bf16_t* __restrict__ Vt,
                                             bf16_t* __restrict__ O) {
  const int lane = threadIdx.x & 31;
  const int t0 = blockIdx.x * 16;
  const int h = blockIdx.y;
  const int b = blockIdx.z;
  const float scale = 0.125f;  // 1/sqrt(HD)
  const bf16_t* Qb = Q + (size_t)b * TSEQ * DIMC + h * HDIM;
  const bf16_t* Kb = K + (size_t)b * TSEQ * DIMC + h * HDIM;
  const bf16_t* Vb = Vt + ((size_t)b * NHEAD + h) * HDIM * TSEQ;  // [HD][T]

  const v16bf aq0 = load_a_frag(Qb, DIMC, t0 + (lane & 15), 0, lane);
  const v16bf aq1 = load_a_frag(Qb, DIMC, t0 + (lane & 15), 32, lane);

  float mst[8], lst[8];
  v8f o[4] = {};
#pragma unroll
  for (int r = 0; r < 8; ++r) { mst[r] = -3.0e38f; lst[r] = 0.f; }

  __shared__ __align__(16) bf16_t pL[16][40];  // 80B row stride, conflict-free
  const int half = lane >> 4, col = lane & 15;
  const int kbEnd = (t0 + 15) >> 5;

  for (int kb = 0; kb <= kbEnd; ++kb) {
    const int key0 = kb * 32;
    // S = Q K^T : K rows are contiguous along d -> K-major B fragments
    v8f s0 = {}, s1 = {};
    {
      v16bf b0 = load_bk_frag(Kb, DIMC, 0, key0, lane);
      v16bf b1 = load_bk_frag(Kb, DIMC, 32, key0, lane);
      s0 = wmma_bf16(aq0, b0, s0);
      s0 = wmma_bf16(aq1, b1, s0);
    }
    {
      v16bf b0 = load_bk_frag(Kb, DIMC, 0, key0 + 16, lane);
      v16bf b1 = load_bk_frag(Kb, DIMC, 32, key0 + 16, lane);
      s1 = wmma_bf16(aq0, b0, s1);
      s1 = wmma_bf16(aq1, b1, s1);
    }
#pragma unroll
    for (int r = 0; r < 8; ++r) {
      const int m = t0 + r + 8 * half;
      float v0 = (key0 + col <= m) ? s0[r] * scale : -INFINITY;
      float v1 = (key0 + 16 + col <= m) ? s1[r] * scale : -INFINITY;
      float vmax = fmaxf(v0, v1);
#pragma unroll
      for (int off = 1; off < 16; off <<= 1) vmax = fmaxf(vmax, __shfl_xor(vmax, off, 32));
      const float mnew = fmaxf(mst[r], vmax);
      const float p0 = __expf(v0 - mnew);
      const float p1 = __expf(v1 - mnew);
      pL[r + 8 * half][col]      = (bf16_t)p0;
      pL[r + 8 * half][16 + col] = (bf16_t)p1;
      float sum = p0 + p1;
#pragma unroll
      for (int off = 1; off < 16; off <<= 1) sum += __shfl_xor(sum, off, 32);
      const float alpha = __expf(mst[r] - mnew);
      lst[r] = lst[r] * alpha + sum;
      mst[r] = mnew;
#pragma unroll
      for (int c = 0; c < 4; ++c) o[c][r] *= alpha;
    }
    __syncthreads();  // single-wave WG: memory fence; LDS pipe is in-order
    v16bf pa = load_a_frag(&pL[0][0], 40, (lane & 15), 0, lane);
#pragma unroll
    for (int c = 0; c < 4; ++c) {
      // B[k=key][n=d] from V^T[d][t]: contiguous along t
      v16bf bv = load_bk_frag(Vb, TSEQ, key0, c * 16, lane);
      o[c] = wmma_bf16(pa, bv, o[c]);
    }
    __syncthreads();
  }

#pragma unroll
  for (int c = 0; c < 4; ++c)
#pragma unroll
    for (int r = 0; r < 8; ++r) {
      const size_t m = t0 + r + 8 * half;
      O[((size_t)b * TSEQ + m) * DIMC + h * HDIM + c * 16 + col] =
          (bf16_t)(o[c][r] / lst[r]);
    }
}

// ---------------------------------------------------------------------------
// Host-side orchestration
// ---------------------------------------------------------------------------
extern "C" void kernel_launch(void* const* d_in, const int* in_sizes, int n_in,
                              void* d_out, int out_size, void* d_ws, size_t ws_size,
                              hipStream_t stream) {
  (void)in_sizes; (void)n_in; (void)out_size; (void)ws_size;
  const float* x  = (const float*)d_in[0];
  // d_in[1] = mask (causal mask applied analytically; unused)
  const float* g1 = (const float*)d_in[2];
  const float* wq = (const float*)d_in[3];
  const float* wk = (const float*)d_in[4];
  const float* wv = (const float*)d_in[5];
  const float* wo = (const float*)d_in[6];
  const float* g2 = (const float*)d_in[7];
  const float* w1 = (const float*)d_in[8];
  const float* w2 = (const float*)d_in[9];
  const float* w3 = (const float*)d_in[10];

  char* ws = (char*)d_ws;
  size_t off = 0;
  auto alloc = [&](size_t bytes) -> void* {
    void* p = ws + off;
    off += (bytes + 255) & ~(size_t)255;
    return p;
  };

  const size_t MT = (size_t)MROWS;
  bf16_t* hb    = (bf16_t*)alloc(MT * DIMC * 2);
  bf16_t* qb    = (bf16_t*)alloc(MT * DIMC * 2);
  bf16_t* kb    = (bf16_t*)alloc(MT * DIMC * 2);
  bf16_t* vtb   = (bf16_t*)alloc(MT * DIMC * 2);  // [B,NH,HD,T]
  bf16_t* ob    = (bf16_t*)alloc(MT * DIMC * 2);
  bf16_t* h2b   = (bf16_t*)alloc(MT * DIMC * 2);
  bf16_t* gateb = (bf16_t*)alloc(MT * FFDIM * 2);
  float*  x1    = (float*) alloc(MT * DIMC * 4);
  bf16_t* wqT = (bf16_t*)alloc((size_t)DIMC * DIMC * 2);
  bf16_t* wkT = (bf16_t*)alloc((size_t)DIMC * DIMC * 2);
  bf16_t* wvT = (bf16_t*)alloc((size_t)DIMC * DIMC * 2);
  bf16_t* woT = (bf16_t*)alloc((size_t)DIMC * DIMC * 2);
  bf16_t* w1T = (bf16_t*)alloc((size_t)DIMC * FFDIM * 2);
  bf16_t* w3T = (bf16_t*)alloc((size_t)DIMC * FFDIM * 2);
  bf16_t* w2T = (bf16_t*)alloc((size_t)FFDIM * DIMC * 2);

  // 1) weights -> bf16, transposed to K-major [N,K]
  const int nw1 = DIMC * DIMC;
  const int nw4 = DIMC * FFDIM;
  k_conv_transpose<<<(nw1 + 255) / 256, 256, 0, stream>>>(wq, wqT, DIMC, DIMC);
  k_conv_transpose<<<(nw1 + 255) / 256, 256, 0, stream>>>(wk, wkT, DIMC, DIMC);
  k_conv_transpose<<<(nw1 + 255) / 256, 256, 0, stream>>>(wv, wvT, DIMC, DIMC);
  k_conv_transpose<<<(nw1 + 255) / 256, 256, 0, stream>>>(wo, woT, DIMC, DIMC);
  k_conv_transpose<<<(nw4 + 255) / 256, 256, 0, stream>>>(w1, w1T, DIMC, FFDIM);
  k_conv_transpose<<<(nw4 + 255) / 256, 256, 0, stream>>>(w3, w3T, DIMC, FFDIM);
  k_conv_transpose<<<(nw4 + 255) / 256, 256, 0, stream>>>(w2, w2T, FFDIM, DIMC);

  // 2) h = rmsnorm(x, g1)
  k_rmsnorm<<<MROWS, 256, 0, stream>>>(x, g1, hb);

  // 3) q,k = h @ {wq,wk};  v = h @ wv scattered into [B,NH,HD,T]
  dim3 gCC(MROWS / 128, DIMC / 64);
  k_gemm<0><<<gCC, 256, 0, stream>>>(hb, wqT, nullptr, qb, MROWS, DIMC, DIMC);
  k_gemm<0><<<gCC, 256, 0, stream>>>(hb, wkT, nullptr, kb, MROWS, DIMC, DIMC);
  k_gemm<2><<<gCC, 256, 0, stream>>>(hb, wvT, nullptr, vtb, MROWS, DIMC, DIMC);

  // 4) RoPE in-place on q, k
  k_rope<<<(BSZ * TSEQ * NHEAD * 32) / 256, 256, 0, stream>>>(qb, kb);

  // 5) causal flash attention -> ob
  dim3 gAtt(TSEQ / 16, NHEAD, BSZ);
  k_attn<<<gAtt, 32, 0, stream>>>(qb, kb, vtb, ob);

  // 6) x1 = x + ob @ wo
  k_gemm<1><<<gCC, 256, 0, stream>>>(ob, woT, x, x1, MROWS, DIMC, DIMC);

  // 7) h2 = rmsnorm(x1, g2)
  k_rmsnorm<<<MROWS, 256, 0, stream>>>(x1, g2, h2b);

  // 8) gate = silu(h2@w1) * (h2@w3)
  dim3 gMLP(MROWS / 64, FFDIM / 64);
  k_gemm_swiglu<<<gMLP, 128, 0, stream>>>(h2b, w1T, w3T, gateb, MROWS, FFDIM, DIMC);

  // 9) out = x1 + gate @ w2
  k_gemm<1><<<gCC, 256, 0, stream>>>(gateb, w2T, x1, (float*)d_out, MROWS, DIMC, FFDIM);
}
